// GroupedRecurrentMultiHeadAttention_9337258902471
// MI455X (gfx1250) — matrix operand order repaired and checked by reading, not compiled, not run on hardware
//
#include <hip/hip_runtime.h>

#define DM   2048
#define DKV  512
#define BATCH 2
#define SEQ  2048
#define ROWS (BATCH*SEQ)   // 4096
#define DH   64

typedef __attribute__((ext_vector_type(16))) __bf16 v16bf;
typedef __attribute__((ext_vector_type(8)))  float  v8f;
typedef __attribute__((ext_vector_type(4)))  unsigned v4u;
typedef __attribute__((ext_vector_type(8)))  int    v8i;
typedef __attribute__((ext_vector_type(4)))  int    v4i;

union Frag {
  uint4 q[2];
  v16bf v;
};

__device__ __forceinline__ unsigned short f2bf(float f) {
  union { float f; unsigned u; } a; a.f = f;
  unsigned r = a.u + 0x7FFFu + ((a.u >> 16) & 1u);   // round-to-nearest-even
  return (unsigned short)(r >> 16);
}
__device__ __forceinline__ float bf2f(unsigned short h) {
  union { unsigned u; float f; } a; a.u = ((unsigned)h) << 16;
  return a.f;
}
__device__ __forceinline__ v8f wmma_bf16(v16bf a, v16bf b, v8f c) {
  return __builtin_amdgcn_wmma_f32_16x16x32_bf16(false, a, false, b, (short)0, c,
                                                 false, false);
}

// ------------------------------------------------------------------ TDM issue
// 2-D tile load via Tensor Data Mover.  D# layout per CDNA5 ISA ch.8.
__device__ __forceinline__ void tdm_issue_2d(unsigned lds_byte_addr,
                                             const void* gptr,
                                             unsigned tensor_d0,
                                             unsigned tensor_d1,
                                             unsigned tile_d0, unsigned tile_d1,
                                             unsigned long long stride0) {
  unsigned long long ga = (unsigned long long)(size_t)gptr;
  v4u g0;
  g0.x = 1u;                                            // count=1, user mode
  g0.y = lds_byte_addr;                                 // lds_addr
  g0.z = (unsigned)ga;                                  // global_addr[31:0]
  g0.w = (unsigned)((ga >> 32) & 0x01FFFFFFu) | 0x80000000u;  // ga[56:32]|type=2
  v8i g1;
  g1[0] = 0x00010000;                                   // data_size=1 (2 bytes)
  g1[1] = (int)((tensor_d0 & 0xFFFFu) << 16);           // abar=0 | dim0 lo
  g1[2] = (int)((tensor_d0 >> 16) | ((tensor_d1 & 0xFFFFu) << 16));
  g1[3] = (int)((tensor_d1 >> 16) | (tile_d0 << 16));
  g1[4] = (int)(tile_d1 & 0xFFFFu);                     // tile_dim2 = 0
  g1[5] = (int)(stride0 & 0xFFFFFFFFu);
  g1[6] = (int)((stride0 >> 32) & 0xFFFFu);             // dim1_stride unused (2D)
  g1[7] = 0;
  v4i gz = {0, 0, 0, 0};
#if __clang_major__ >= 23
  v8i gz8 = {0, 0, 0, 0, 0, 0, 0, 0};
  __builtin_amdgcn_tensor_load_to_lds(g0, g1, gz, gz, gz8, 0);
#else
  __builtin_amdgcn_tensor_load_to_lds(g0, g1, gz, gz, 0);
#endif
}

// ---------------------------------------------------------------- pack f32->bf16
__global__ void pack_bf16_k(const float* __restrict__ src,
                            unsigned short* __restrict__ dst, int n) {
  for (int i = blockIdx.x * blockDim.x + threadIdx.x; i < n;
       i += gridDim.x * blockDim.x)
    dst[i] = f2bf(src[i]);
}

// memory[k,n] (f32) -> memT[n,k] (bf16), 512x512
__global__ void pack_memT_k(const float* __restrict__ mem,
                            unsigned short* __restrict__ dst) {
  int n = blockIdx.x;
  for (int k = threadIdx.x; k < DKV; k += blockDim.x)
    dst[n * DKV + k] = f2bf(mem[k * DKV + n]);
}

// ------------------------------------------------------------------ GEMM core
// Block tile 128(M) x 64(N), K super-steps of 256, TDM double-buffered LDS.
// LDS (bytes): A0 @ 0 (64KB), A1 @ 65536, B0 @ 131072 (32KB), B1 @ 163840.
// Inner loop register-double-buffers fragments so ds_loads overlap WMMAs.
template <int KTOT>
__device__ __forceinline__ void gemm_mainloop(const unsigned short* __restrict__ A,
                                              const unsigned short* __restrict__ Bm,
                                              unsigned aRows, unsigned bRows,
                                              int mBase, int nBase,
                                              unsigned short* sm, v8f acc[4]) {
  const int SUP = KTOT / 256;
  int wave = threadIdx.x >> 5, lane = threadIdx.x & 31;
  int lr = lane & 15, hi = lane >> 4;
  unsigned ldsBase = (unsigned)(size_t)sm;   // LDS aperture: offset in addr[31:0]

  if (wave == 0) {
    tdm_issue_2d(ldsBase, A + (size_t)mBase * KTOT, KTOT, aRows, 256, 128, KTOT);
    tdm_issue_2d(ldsBase + 131072u, Bm + (size_t)nBase * KTOT, KTOT, bRows,
                 256, 64, KTOT);
  }
  for (int s = 0; s < SUP; ++s) {
    unsigned aoff = (s & 1) ? 65536u : 0u;
    unsigned boff = (s & 1) ? 163840u : 131072u;
    if (wave == 0) __builtin_amdgcn_s_wait_tensorcnt(0);  // step s landed
    __syncthreads();
    if (wave == 0 && s + 1 < SUP) {                        // prefetch step s+1
      int k0 = (s + 1) * 256;
      unsigned aoff2 = ((s + 1) & 1) ? 65536u : 0u;
      unsigned boff2 = ((s + 1) & 1) ? 163840u : 131072u;
      tdm_issue_2d(ldsBase + aoff2, A + (size_t)mBase * KTOT + k0, KTOT, aRows,
                   256, 128, KTOT);
      tdm_issue_2d(ldsBase + boff2, Bm + (size_t)nBase * KTOT + k0, KTOT, bRows,
                   256, 64, KTOT);
    }
    const unsigned short* As =
        (const unsigned short*)((char*)sm + aoff) +
        (size_t)(wave * 16 + lr) * 256 + hi * 8;
    const unsigned short* Bs =
        (const unsigned short*)((char*)sm + boff) +
        (size_t)lr * 256 + hi * 8;

    Frag af[2];
    Frag bfr[2][4];
    af[0].q[0] = *(const uint4*)(As);
    af[0].q[1] = *(const uint4*)(As + 16);
#pragma unroll
    for (int j = 0; j < 4; ++j) {
      const unsigned short* bp = Bs + j * 4096;
      bfr[0][j].q[0] = *(const uint4*)bp;
      bfr[0][j].q[1] = *(const uint4*)(bp + 16);
    }
#pragma unroll
    for (int kc = 0; kc < 8; ++kc) {
      int cur = kc & 1, nxt = cur ^ 1;
      if (kc < 7) {                      // preload sub-step kc+1
        const unsigned short* An = As + (kc + 1) * 32;
        af[nxt].q[0] = *(const uint4*)An;
        af[nxt].q[1] = *(const uint4*)(An + 16);
#pragma unroll
        for (int j = 0; j < 4; ++j) {
          const unsigned short* bp = Bs + j * 4096 + (kc + 1) * 32;
          bfr[nxt][j].q[0] = *(const uint4*)bp;
          bfr[nxt][j].q[1] = *(const uint4*)(bp + 16);
        }
      }
#pragma unroll
      for (int j = 0; j < 4; ++j)
        acc[j] = wmma_bf16(af[cur].v, bfr[cur][j].v, acc[j]);
    }
  }
}

// ---------------------------------------------------------------- QKV GEMM
__global__ void __launch_bounds__(256)
gemm_qkv_k(const unsigned short* __restrict__ xb,
           const unsigned short* __restrict__ wb,
           unsigned short* __restrict__ qb,
           unsigned short* __restrict__ sqb,
           unsigned short* __restrict__ kb,
           unsigned short* __restrict__ vt) {
  extern __shared__ __align__(16) unsigned short sm[];
  int wave = threadIdx.x >> 5, lane = threadIdx.x & 31;
  int lr = lane & 15, hi = lane >> 4;
  int mBase = blockIdx.y * 128, nBase = blockIdx.x * 64;
  const v8f zero8 = {0.f, 0.f, 0.f, 0.f, 0.f, 0.f, 0.f, 0.f};
  v8f acc[4];
#pragma unroll
  for (int j = 0; j < 4; ++j) acc[j] = zero8;

  gemm_mainloop<DM>(xb, wb, ROWS, 3072, mBase, nBase, sm, acc);

#pragma unroll
  for (int j = 0; j < 4; ++j) {
    int col = nBase + j * 16 + lr;
#pragma unroll
    for (int i = 0; i < 8; ++i) {
      int row = mBase + wave * 16 + i + 8 * hi;
      float val = acc[j][i];
      if (col < DM) {
        qb[(size_t)row * DM + col] = f2bf(val);
        float e = val > 0.f ? val : (__expf(val) - 1.f);   // elu
        sqb[(size_t)row * DM + col] = f2bf(e);
      } else if (col < DM + DKV) {
        kb[(size_t)row * DKV + (col - DM)] = f2bf(val);
      } else {
        int c = col - (DM + DKV);
        int b = row >> 11, s = row & (SEQ - 1);
        vt[((size_t)(b * DKV + c)) * SEQ + s] = f2bf(val);   // V transposed
      }
    }
  }
}

// ---------------------------------------------------------------- a_mem GEMM
__global__ void __launch_bounds__(256)
gemm_mem_k(const unsigned short* __restrict__ sqb,
           const unsigned short* __restrict__ memT,
           const float* __restrict__ sqsum,
           const float* __restrict__ memw,
           float* __restrict__ out) {
  extern __shared__ __align__(16) unsigned short sm[];
  int wave = threadIdx.x >> 5, lane = threadIdx.x & 31;
  int lr = lane & 15, hi = lane >> 4;
  int mBase = blockIdx.y * 128, nBase = blockIdx.x * 64;
  const v8f zero8 = {0.f, 0.f, 0.f, 0.f, 0.f, 0.f, 0.f, 0.f};
  v8f acc[4];
#pragma unroll
  for (int j = 0; j < 4; ++j) acc[j] = zero8;

  gemm_mainloop<DKV>(sqb, memT, 16384u, DKV, mBase, nBase, sm, acc);

  float w = 1.f / (1.f + __expf(-memw[0]));
#pragma unroll
  for (int j = 0; j < 4; ++j)
#pragma unroll
    for (int i = 0; i < 8; ++i) {
      int row = mBase + wave * 16 + i + 8 * hi;   // sigma_q row in [16384]
      int col = nBase + j * 16 + lr;              // 0..511
      float am = acc[j][i] / sqsum[row];
      int orow = row >> 2, qg = row & 3;
      size_t oi = (size_t)orow * DM + qg * DKV + col;
      out[oi] += w * am;                          // stream-ordered after attn_k
    }
}

// ---------------------------------------------------------------- sigma_q @ memory_norm
__global__ void rowsum_k(const unsigned short* __restrict__ sqb,
                         const float* __restrict__ norm,
                         float* __restrict__ sqsum) {
  int wave = threadIdx.x >> 5, lane = threadIdx.x & 31;
  int row = blockIdx.x * 8 + wave;           // 16384 rows of 512
  const unsigned short* p = sqb + (size_t)row * DKV;
  float s = 0.f;
  for (int c = lane; c < DKV; c += 32) s += bf2f(p[c]) * norm[c];
#pragma unroll
  for (int off = 16; off > 0; off >>= 1) s += __shfl_xor(s, off, 32);
  if (lane == 0) sqsum[row] = s;
}

// ---------------------------------------------------------------- flash attention
// grid: x = 32 (64 query rows per block), y = 64 combos (b*32 + qg*8 + h)
__global__ void __launch_bounds__(128)
attn_k(const unsigned short* __restrict__ qb,
       const unsigned short* __restrict__ kb,
       const unsigned short* __restrict__ vt,
       const float* __restrict__ memw,
       float* __restrict__ out) {
  __shared__ __align__(16) unsigned short pls[4][16 * 32];
  int wave = threadIdx.x >> 5, lane = threadIdx.x & 31;
  int lr = lane & 15, hi = lane >> 4;
  int combo = blockIdx.y;
  int b = combo >> 5, qg = (combo >> 3) & 3, h = combo & 7;
  int s0 = blockIdx.x * 64 + wave * 16;
  size_t rbase = (size_t)b * SEQ;
  const float scale = 0.125f;   // 1/sqrt(64)
  const v8f zero8 = {0.f, 0.f, 0.f, 0.f, 0.f, 0.f, 0.f, 0.f};

  Frag qa[2];
  const unsigned short* qp = qb + (rbase + s0 + lr) * DM + qg * DKV + h * DH;
#pragma unroll
  for (int kk = 0; kk < 2; ++kk) {
    const unsigned short* p = qp + kk * 32 + hi * 8;
    qa[kk].q[0] = *(const uint4*)p;
    qa[kk].q[1] = *(const uint4*)(p + 16);
  }

  v8f o[4];
#pragma unroll
  for (int j = 0; j < 4; ++j) o[j] = zero8;
  float m_run[8], l_run[8];
#pragma unroll
  for (int i = 0; i < 8; ++i) { m_run[i] = -1e30f; l_run[i] = 0.f; }

  for (int t0 = 0; t0 < SEQ; t0 += 32) {
    // load all K and V fragments of this chunk up front: one wait, long bursts;
    // V loads overlap the QK WMMAs and softmax below.
    Frag kf[2][2];
#pragma unroll
    for (int tt = 0; tt < 2; ++tt) {
      const unsigned short* kp = kb + (rbase + t0 + tt * 16 + lr) * DKV + h * DH;
#pragma unroll
      for (int kk = 0; kk < 2; ++kk) {
        const unsigned short* p = kp + kk * 32 + hi * 8;
        kf[tt][kk].q[0] = *(const uint4*)p;
        kf[tt][kk].q[1] = *(const uint4*)(p + 16);
      }
    }
    Frag vf[4];
    const unsigned short* vp =
        vt + ((size_t)(b * DKV + h * DH)) * SEQ + t0 + hi * 8;
#pragma unroll
    for (int j = 0; j < 4; ++j) {
      const unsigned short* p = vp + (size_t)(j * 16 + lr) * SEQ;
      vf[j].q[0] = *(const uint4*)p;
      vf[j].q[1] = *(const uint4*)(p + 16);
    }
    if (t0 + 32 < SEQ) {   // cache prefetch of next chunk (global_prefetch_b8)
      __builtin_prefetch(kb + (rbase + t0 + 32 + lr) * DKV + h * DH, 0, 0);
      __builtin_prefetch(vp + 32, 0, 0);
    }

    v8f sa0 = zero8, sa1 = zero8;
#pragma unroll
    for (int kk = 0; kk < 2; ++kk) sa0 = wmma_bf16(qa[kk].v, kf[0][kk].v, sa0);
#pragma unroll
    for (int kk = 0; kk < 2; ++kk) sa1 = wmma_bf16(qa[kk].v, kf[1][kk].v, sa1);

    float mc[8], rs[8], alpha[8];
#pragma unroll
    for (int i = 0; i < 8; ++i) {
      float a = sa0[i] * scale, c = sa1[i] * scale;
      sa0[i] = a; sa1[i] = c;
      mc[i] = fmaxf(a, c);
    }
#pragma unroll
    for (int off = 1; off < 16; off <<= 1)
#pragma unroll
      for (int i = 0; i < 8; ++i)
        mc[i] = fmaxf(mc[i], __shfl_xor(mc[i], off, 32));
#pragma unroll
    for (int i = 0; i < 8; ++i) {
      float mn = fmaxf(m_run[i], mc[i]);
      alpha[i] = __expf(m_run[i] - mn);
      m_run[i] = mn;
      float p0 = __expf(sa0[i] - mn);
      float p1 = __expf(sa1[i] - mn);
      sa0[i] = p0; sa1[i] = p1;
      rs[i] = p0 + p1;
    }
#pragma unroll
    for (int off = 1; off < 16; off <<= 1)
#pragma unroll
      for (int i = 0; i < 8; ++i) rs[i] += __shfl_xor(rs[i], off, 32);
#pragma unroll
    for (int i = 0; i < 8; ++i) l_run[i] = l_run[i] * alpha[i] + rs[i];
#pragma unroll
    for (int j = 0; j < 4; ++j)
#pragma unroll
      for (int i = 0; i < 8; ++i) o[j][i] *= alpha[i];

    // P: C-layout f32 -> A-layout bf16 via LDS round trip
    __syncthreads();   // WAR vs previous iteration's frag loads
#pragma unroll
    for (int i = 0; i < 8; ++i) {
      pls[wave][(i + 8 * hi) * 32 + lr]      = f2bf(sa0[i]);
      pls[wave][(i + 8 * hi) * 32 + 16 + lr] = f2bf(sa1[i]);
    }
    __syncthreads();
    Frag pf;
    const unsigned short* lp = &pls[wave][lr * 32 + hi * 8];
    pf.q[0] = *(const uint4*)lp;
    pf.q[1] = *(const uint4*)(lp + 16);

#pragma unroll
    for (int j = 0; j < 4; ++j) o[j] = wmma_bf16(pf.v, vf[j].v, o[j]);
  }

  float w = 1.f / (1.f + __expf(-memw[0]));
  float g = 1.f - w;
#pragma unroll
  for (int j = 0; j < 4; ++j)
#pragma unroll
    for (int i = 0; i < 8; ++i) {
      int row = s0 + i + 8 * hi;
      int col = qg * DKV + h * DH + j * 16 + lr;
      out[(rbase + row) * DM + col] = g * o[j][i] / l_run[i];
    }
}

// ---------------------------------------------------------------- launch
extern "C" void kernel_launch(void* const* d_in, const int* in_sizes, int n_in,
                              void* d_out, int out_size, void* d_ws,
                              size_t ws_size, hipStream_t stream) {
  (void)in_sizes; (void)n_in; (void)out_size; (void)ws_size;
  const float* x       = (const float*)d_in[0];
  const float* Wq      = (const float*)d_in[1];
  const float* Wk      = (const float*)d_in[2];
  const float* Wv      = (const float*)d_in[3];
  const float* mem     = (const float*)d_in[4];
  const float* memnorm = (const float*)d_in[5];
  const float* memw    = (const float*)d_in[6];
  float* out = (float*)d_out;
  char* ws = (char*)d_ws;

  // workspace layout (bytes)
  unsigned short* xb   = (unsigned short*)(ws);             // 4096x2048 bf16
  unsigned short* Wb   = (unsigned short*)(ws + 16777216);  // 3072x2048 bf16
  unsigned short* qb   = (unsigned short*)(ws + 29360128);  // 4096x2048 bf16
  unsigned short* sqb  = (unsigned short*)(ws + 46137344);  // elu(q) bf16
  unsigned short* kbuf = (unsigned short*)(ws + 62914560);  // 4096x512 bf16
  unsigned short* vt   = (unsigned short*)(ws + 67108864);  // 2x512x2048 bf16 (V^T)
  unsigned short* memT = (unsigned short*)(ws + 71303168);  // 512x512 bf16
  float*          sqsum = (float*)(ws + 71827456);          // 16384 f32

  const size_t GEMM_LDS = 196608;   // 2x(64KB A + 32KB B)

  pack_bf16_k<<<4096, 256, 0, stream>>>(x, xb, ROWS * DM);
  pack_bf16_k<<<4096, 256, 0, stream>>>(Wq, Wb, DM * DM);
  pack_bf16_k<<<2048, 256, 0, stream>>>(Wk, Wb + (size_t)DM * DM, DKV * DM);
  pack_bf16_k<<<2048, 256, 0, stream>>>(Wv, Wb + (size_t)(DM + DKV) * DM,
                                        DKV * DM);
  pack_memT_k<<<512, 256, 0, stream>>>(mem, memT);

  gemm_qkv_k<<<dim3(48, 32), 256, GEMM_LDS, stream>>>(xb, Wb, qb, sqb, kbuf, vt);
  rowsum_k<<<2048, 256, 0, stream>>>(sqb, memnorm, sqsum);
  attn_k<<<dim3(32, 64), 128, 0, stream>>>(qb, kbuf, vt, memw, out);
  gemm_mem_k<<<dim3(8, 128), 256, GEMM_LDS, stream>>>(sqb, memT, sqsum, memw, out);
}
